// RoutingTransformerModule_44332652429355
// MI455X (gfx1250) — compile-verified, matrix operand order checked by reading, not run on hardware
//
#include <hip/hip_runtime.h>
#include <hip/hip_bf16.h>
#include <math.h>

// ---------------- problem constants ----------------
#define B_    4
#define S_    4096
#define E_    512
#define H_    8
#define L_    4
#define A_    3
#define NVOC  17          // NV+1
#define NPOS  65          // 2^TD + 1
#define WIN_  64
#define C_    64          // S/WIN
#define D_    64          // E/H
#define FF_   2048        // 4*E
#define MTOK  (B_*S_)

typedef __attribute__((ext_vector_type(16))) _Float16 v16h;
typedef __attribute__((ext_vector_type(8)))  float    v8f;

// CDNA5 async global->LDS copy path (ASYNCcnt), guarded so a missing builtin
// falls back to plain vectorized LDS staging.
#if defined(__AMDGCN__) && __has_builtin(__builtin_amdgcn_global_load_async_to_lds_b128) && __has_builtin(__builtin_amdgcn_s_wait_asynccnt)
#define USE_ASYNC_LDS 1
#define AS1 __attribute__((address_space(1)))
#define AS3 __attribute__((address_space(3)))
// exact parameter type per hipcc diagnostic: gcc-style int4 vector pointer
typedef int v4i_b __attribute__((vector_size(16)));
#else
#define USE_ASYNC_LDS 0
#endif

// ---------------- WMMA helpers ----------------
__device__ __forceinline__ v8f wmma16x16x32(v16h a, v16h b, v8f c) {
  return __builtin_amdgcn_wmma_f32_16x16x32_f16(
      /*neg_a=*/false, a, /*neg_b=*/false, b,
      /*c_mod=*/(short)0, c, /*reuse_a=*/false, /*reuse_b=*/false);
}

// A fragment: 16x32 f16 tile, row-major (M-major, K contiguous), leading dim lda.
// Per lane this is two contiguous 8-half runs -> 2x ds_load_b128.
__device__ __forceinline__ v16h ldsA(const _Float16* p, int lda, int lane) {
  int row = lane & 15;
  int hi  = (lane >> 4) ? 8 : 0;
  v16h a;
#pragma unroll
  for (int j = 0; j < 16; ++j) {
    int vg = j >> 1;
    int kk = ((vg & 4) ? 16 : 0) + hi + 2 * (vg & 3) + (j & 1);
    a[j] = p[row * lda + kk];
  }
  return a;
}

// B fragment from an N-major (transposed) tile: pT[n][k], leading dim ldb.
// Same contiguous-run property as A -> 2x ds_load_b128 per fragment.
__device__ __forceinline__ v16h ldsBT(const _Float16* pT, int ldb, int lane) {
  int col = lane & 15;
  int hi  = (lane >> 4) ? 8 : 0;
  v16h b;
#pragma unroll
  for (int j = 0; j < 16; ++j) {
    int vg = j >> 1;
    int kk = ((vg & 4) ? 16 : 0) + hi + 2 * (vg & 3) + (j & 1);
    b[j] = pT[col * ldb + kk];
  }
  return b;
}

__device__ __forceinline__ float gelu_f(float x) {
  const float k0 = 0.7978845608028654f, k1 = 0.044715f;
  return 0.5f * x * (1.f + tanhf(k0 * (x + k1 * x * x * x)));
}

// ---------------- f32 -> f16 convert ----------------
__global__ void __launch_bounds__(256)
f32_to_f16_kernel(const float* __restrict__ src, _Float16* __restrict__ dst, int n) {
  int i = blockIdx.x * 256 + threadIdx.x;
  if (i < n) dst[i] = (_Float16)src[i];
}

// ---------------- embedding + shift-right ----------------
__global__ void __launch_bounds__(256)
embed_kernel(const int* __restrict__ value, const int* __restrict__ depth,
             const int* __restrict__ pos, const float* __restrict__ sos,
             const float* __restrict__ tok_emb, const float* __restrict__ depth_emb,
             const float* __restrict__ pos_emb, float* __restrict__ x) {
  int t = blockIdx.x;               // b*S + s
  int s = t % S_;
  int tid = threadIdx.x;
  float r0, r1;
  if (s == 0) {
    r0 = sos[tid]; r1 = sos[tid + 256];
  } else {
    int sp = t - 1;                 // (b, s-1)
    int v = value[sp], d = depth[sp];
    r0 = tok_emb[(size_t)v * E_ + tid]       + depth_emb[(size_t)d * E_ + tid];
    r1 = tok_emb[(size_t)v * E_ + tid + 256] + depth_emb[(size_t)d * E_ + tid + 256];
#pragma unroll
    for (int a = 0; a < A_; ++a) {
      int p = pos[(size_t)sp * A_ + a];
      const float* pe = pos_emb + ((size_t)a * NPOS + p) * E_;
      r0 += pe[tid]; r1 += pe[tid + 256];
    }
  }
  x[(size_t)t * E_ + tid]       = r0;
  x[(size_t)t * E_ + tid + 256] = r1;
}

// ---------------- layernorm (one block per token) -> f16 ----------------
__global__ void __launch_bounds__(256)
layernorm_kernel(const float* __restrict__ x, const float* __restrict__ sc,
                 const float* __restrict__ bi, _Float16* __restrict__ o16) {
  int t = blockIdx.x, tid = threadIdx.x;
  const float* xp = x + (size_t)t * E_;
  float v0 = xp[tid], v1 = xp[tid + 256];
  __shared__ float red[256];
  red[tid] = v0 + v1; __syncthreads();
  for (int st = 128; st > 0; st >>= 1) { if (tid < st) red[tid] += red[tid + st]; __syncthreads(); }
  float mean = red[0] * (1.f / E_);
  __syncthreads();
  float d0 = v0 - mean, d1 = v1 - mean;
  red[tid] = d0 * d0 + d1 * d1; __syncthreads();
  for (int st = 128; st > 0; st >>= 1) { if (tid < st) red[tid] += red[tid + st]; __syncthreads(); }
  float rstd = rsqrtf(red[0] * (1.f / E_) + 1e-5f);
  o16[(size_t)t * E_ + tid]       = (_Float16)(d0 * rstd * sc[tid]       + bi[tid]);
  o16[(size_t)t * E_ + tid + 256] = (_Float16)(d1 * rstd * sc[tid + 256] + bi[tid + 256]);
}

// ---------------- generic WMMA GEMM (f16 in, f32 or f16 out) ----------------
// C[M,N] = epi(A[M,K] @ B); B is f16 [K,N], or [N,K] if TRANSB.
template <bool TRANSB, bool GELU, bool RESID, bool BIAS, bool OUTH>
__global__ void __launch_bounds__(256)
gemm_wmma_kernel(const _Float16* __restrict__ A, const _Float16* __restrict__ Bm,
                 const float* __restrict__ bias, const float* __restrict__ R,
                 void* __restrict__ Co_, int M, int N, int K) {
  const int TM = 64, TN = 64, TK = 64, LDT = TK + 8;
  __shared__ _Float16 As[TM * LDT];    // [m][k], K contiguous
  __shared__ _Float16 BsT[TN * LDT];   // [n][k], K contiguous
  int tid = threadIdx.x, lane = tid & 31, wave = tid >> 5;
  int wr = wave & 3, wc = wave >> 2;
  int row0 = blockIdx.y * TM, col0 = blockIdx.x * TN;
  v8f acc0 = {}, acc1 = {};
  for (int k0 = 0; k0 < K; k0 += TK) {
    // ---- A tile: 64x64 halves, 16B chunks ----
#if USE_ASYNC_LDS
    for (int i = tid; i < (TM * TK) / 8; i += 256) {
      int r = i >> 3, c = (i & 7) * 8;
      __builtin_amdgcn_global_load_async_to_lds_b128(
          (AS1 v4i_b*)(A + (size_t)(row0 + r) * K + k0 + c),
          (AS3 v4i_b*)(As + r * LDT + c), 0, 0);
    }
#else
    for (int i = tid; i < (TM * TK) / 8; i += 256) {
      int r = i >> 3, c = (i & 7) * 8;
      *(uint4*)&As[r * LDT + c] =
          *(const uint4*)&A[(size_t)(row0 + r) * K + k0 + c];
    }
#endif
    // ---- B tile into BsT[n][k] ----
    if (TRANSB) {
      // B is [N,K]: rows already K-contiguous (guard N, e.g. head N=17)
      for (int i = tid; i < (TN * TK) / 8; i += 256) {
        int n = i >> 3, c = (i & 7) * 8;
        int gn = col0 + n;
        uint4 val = {0u, 0u, 0u, 0u};
        if (gn < N) val = *(const uint4*)&Bm[(size_t)gn * K + k0 + c];
        *(uint4*)&BsT[n * LDT + c] = val;
      }
    } else {
      // B is [K,N] (N multiple of 64 on all such paths): coalesced row reads,
      // scalar transpose into LDS.
      for (int i = tid; i < (TK * TN) / 8; i += 256) {
        int k = i >> 3, nc = (i & 7) * 8;
        uint4 val = *(const uint4*)&Bm[(size_t)(k0 + k) * N + col0 + nc];
        const _Float16* hv = (const _Float16*)&val;
#pragma unroll
        for (int j = 0; j < 8; ++j) BsT[(nc + j) * LDT + k] = hv[j];
      }
      if (k0 + TK < K)
        __builtin_prefetch(&Bm[(size_t)(k0 + TK + (tid >> 3)) * N + col0], 0, 1);
    }
#if USE_ASYNC_LDS
    __builtin_amdgcn_s_wait_asynccnt(0);
#endif
    __syncthreads();
#pragma unroll
    for (int kc = 0; kc < TK; kc += 32) {
      v16h af = ldsA(&As[(wr * 16) * LDT + kc], LDT, lane);
      v16h b0 = ldsBT(&BsT[(wc * 32) * LDT + kc], LDT, lane);
      v16h b1 = ldsBT(&BsT[(wc * 32 + 16) * LDT + kc], LDT, lane);
      acc0 = wmma16x16x32(af, b0, acc0);
      acc1 = wmma16x16x32(af, b1, acc1);
    }
    __syncthreads();
  }
  int colA = lane & 15, rb = (lane >> 4) * 8;
#pragma unroll
  for (int i = 0; i < 8; ++i) {
    int r = row0 + wr * 16 + rb + i;
#pragma unroll
    for (int s = 0; s < 2; ++s) {
      int cidx = col0 + wc * 32 + s * 16 + colA;
      if (r < M && cidx < N) {
        float v = (s == 0) ? acc0[i] : acc1[i];
        if (BIAS)  v += bias[cidx];
        if (GELU)  v = gelu_f(v);
        if (RESID) v += R[(size_t)r * N + cidx];
        if (OUTH) ((_Float16*)Co_)[(size_t)r * N + cidx] = (_Float16)v;
        else      ((float*)Co_)[(size_t)r * N + cidx] = v;
      }
    }
  }
}

// ---------------- q norms (one wave per (token, head)) ----------------
__global__ void __launch_bounds__(256)
qnorm_kernel(const _Float16* __restrict__ q, float* __restrict__ qn) {
  int wave = threadIdx.x >> 5, lane = threadIdx.x & 31;
  int pair = blockIdx.x * 8 + wave;          // token*H + h
  int tok = pair >> 3, h = pair & 7;
  const _Float16* qp = q + (size_t)tok * E_ + h * D_;
  float a = (float)qp[lane], b = (float)qp[lane + 32];
  float ss = a * a + b * b;
#pragma unroll
  for (int m = 16; m; m >>= 1) ss += __shfl_xor(ss, m, 32);
  if (lane == 0) qn[pair] = sqrtf(ss);
}

// ---------------- normalized centroids ----------------
__global__ void __launch_bounds__(64)
mnorm_kernel(const float* __restrict__ means, float* __restrict__ mn) {
  int cidx = blockIdx.x;                      // h*C + c
  int tid = threadIdx.x;
  float v = means[(size_t)cidx * D_ + tid];
  __shared__ float red[64];
  red[tid] = v * v; __syncthreads();
  for (int st = 32; st > 0; st >>= 1) { if (tid < st) red[tid] += red[tid + st]; __syncthreads(); }
  mn[(size_t)cidx * D_ + tid] = v / (sqrtf(red[0]) + 1e-8f);
}

// ---------------- routing: dists + top-WIN per (b,h,c) ----------------
__global__ void __launch_bounds__(256)
route_topk_kernel(const _Float16* __restrict__ q, const float* __restrict__ qn,
                  const float* __restrict__ mn, int* __restrict__ idx) {
  int bid = blockIdx.x;                       // b*H*C + h*C + c
  int c = bid % C_, h = (bid / C_) % H_, b = bid / (H_ * C_);
  int tid = threadIdx.x;
  __shared__ float cm[D_];
  __shared__ float dist[S_];
  __shared__ float rv[256];
  __shared__ int   ri[256];
  if (tid < D_) cm[tid] = mn[((size_t)h * C_ + c) * D_ + tid];
  __syncthreads();
  for (int n = tid; n < S_; n += 256) {
    const _Float16* qp = q + ((size_t)b * S_ + n) * E_ + h * D_;
    float dot = 0.f;
#pragma unroll 8
    for (int d = 0; d < D_; ++d) dot += (float)qp[d] * cm[d];
    dist[n] = dot / (qn[((size_t)b * S_ + n) * H_ + h] + 1e-8f);
  }
  __syncthreads();
  int* op = idx + (size_t)bid * WIN_;
  for (int w = 0; w < WIN_; ++w) {
    float bv = -3.0e38f; int bi = 0;
    for (int n = tid; n < S_; n += 256) {
      float dv = dist[n];
      if (dv > bv) { bv = dv; bi = n; }
    }
    rv[tid] = bv; ri[tid] = bi; __syncthreads();
    for (int st = 128; st > 0; st >>= 1) {
      if (tid < st) {
        if (rv[tid + st] > rv[tid] ||
            (rv[tid + st] == rv[tid] && ri[tid + st] < ri[tid])) {
          rv[tid] = rv[tid + st]; ri[tid] = ri[tid + st];
        }
      }
      __syncthreads();
    }
    if (tid == 0) { op[w] = ri[0]; dist[ri[0]] = -3.0e38f; }
    __syncthreads();
  }
}

// ---------------- bucket attention (WMMA) ----------------
#define LDH 72
#define LDSS 68
__global__ void __launch_bounds__(256)
bucket_attn_kernel(const _Float16* __restrict__ q, const _Float16* __restrict__ k,
                   const _Float16* __restrict__ v, const int* __restrict__ idx,
                   const int* __restrict__ value, float* __restrict__ out,
                   float* __restrict__ cnt) {
  int bid = blockIdx.x;                       // b*H*C + h*C + c
  int h = (bid / C_) % H_, b = bid / (H_ * C_);
  int tid = threadIdx.x, lane = tid & 31, wave = tid >> 5;
  int wr = wave & 3, wc = wave >> 2;
  __shared__ _Float16 qs[64 * LDH];    // [w][d]   (A for scores)
  __shared__ _Float16 ks[64 * LDH];    // [x][d]   (N-major B for scores)
  __shared__ _Float16 vsT[64 * LDH];   // [d][x]   (N-major B for out)
  __shared__ _Float16 ph[64 * LDH];    // [w][x]   (A for out)
  __shared__ float ss[64 * LDSS];
  __shared__ int   toks[64];
  __shared__ float kmv[64];
  if (tid < 64) {
    int t = idx[(size_t)bid * WIN_ + tid];
    toks[tid] = t;
    kmv[tid] = (value[(size_t)b * S_ + t] != 0) ? 1.f : 0.f;
  }
  __syncthreads();
  // q/k rows: 16B vector copies; v scattered transposed
  for (int i = tid; i < 64 * 8; i += 256) {
    int w = i >> 3, c = (i & 7) * 8;
    size_t base = ((size_t)b * S_ + toks[w]) * E_ + h * D_ + c;
    *(uint4*)&qs[w * LDH + c] = *(const uint4*)&q[base];
    *(uint4*)&ks[w * LDH + c] = *(const uint4*)&k[base];
  }
  for (int i = tid; i < 64 * 64; i += 256) {
    int w = i >> 6, d = i & 63;
    vsT[d * LDH + w] = v[((size_t)b * S_ + toks[w]) * E_ + h * D_ + d];
  }
  __syncthreads();
  { // scores = qb @ kb^T
    v8f a0 = {}, a1 = {};
#pragma unroll
    for (int kc = 0; kc < 64; kc += 32) {
      v16h af = ldsA(&qs[(wr * 16) * LDH + kc], LDH, lane);
      v16h b0 = ldsBT(&ks[(wc * 32) * LDH + kc], LDH, lane);
      v16h b1 = ldsBT(&ks[(wc * 32 + 16) * LDH + kc], LDH, lane);
      a0 = wmma16x16x32(af, b0, a0);
      a1 = wmma16x16x32(af, b1, a1);
    }
    int colA = lane & 15, rb = (lane >> 4) * 8;
#pragma unroll
    for (int i = 0; i < 8; ++i) {
      int r = wr * 16 + rb + i;
      ss[r * LDSS + wc * 32 + colA]      = a0[i];
      ss[r * LDSS + wc * 32 + 16 + colA] = a1[i];
    }
  }
  __syncthreads();
  if (tid < 64) {    // masked softmax, one thread per query row
    int r = tid, tr = toks[r];
    float sc[64];
    float mx = -3.4e38f;
    for (int x = 0; x < 64; ++x) {
      bool ok = (tr >= toks[x]) && (kmv[x] > 0.5f);
      float val = ok ? ss[r * LDSS + x] * 0.125f : -1e9f;
      sc[x] = val; mx = fmaxf(mx, val);
    }
    float sum = 0.f;
    for (int x = 0; x < 64; ++x) { float e = __expf(sc[x] - mx); sc[x] = e; sum += e; }
    float inv = 1.f / sum;
    for (int x = 0; x < 64; ++x) ph[r * LDH + x] = (_Float16)(sc[x] * inv);
  }
  __syncthreads();
  { // ob = attn @ vb ; scatter-add
    v8f a0 = {}, a1 = {};
#pragma unroll
    for (int kc = 0; kc < 64; kc += 32) {
      v16h af = ldsA(&ph[(wr * 16) * LDH + kc], LDH, lane);
      v16h b0 = ldsBT(&vsT[(wc * 32) * LDH + kc], LDH, lane);
      v16h b1 = ldsBT(&vsT[(wc * 32 + 16) * LDH + kc], LDH, lane);
      a0 = wmma16x16x32(af, b0, a0);
      a1 = wmma16x16x32(af, b1, a1);
    }
    int colA = lane & 15, rb = (lane >> 4) * 8;
#pragma unroll
    for (int i = 0; i < 8; ++i) {
      int w = wr * 16 + rb + i;
      size_t base = ((size_t)b * S_ + toks[w]) * E_ + h * D_;
      atomicAdd(&out[base + wc * 32 + colA],      a0[i]);
      atomicAdd(&out[base + wc * 32 + 16 + colA], a1[i]);
    }
  }
  if (tid < 64) atomicAdd(&cnt[((size_t)b * H_ + h) * S_ + toks[tid]], 1.0f);
}

// ---------------- scatter-mean finalize -> f16 ----------------
__global__ void __launch_bounds__(256)
attn_norm_kernel(const float* __restrict__ out, const float* __restrict__ cnt,
                 _Float16* __restrict__ o16) {
  int t = blockIdx.x;                         // token index b*S+n
  int b = t / S_, n = t % S_;
  int tid = threadIdx.x;
  for (int e = tid; e < E_; e += 256) {
    int h = e >> 6;
    float c = cnt[((size_t)b * H_ + h) * S_ + n];
    o16[(size_t)t * E_ + e] = (_Float16)(out[(size_t)t * E_ + e] / fmaxf(c, 1.f));
  }
}

// ---------------- host orchestration ----------------
extern "C" void kernel_launch(void* const* d_in, const int* in_sizes, int n_in,
                              void* d_out, int out_size, void* d_ws, size_t ws_size,
                              hipStream_t stream) {
  (void)in_sizes; (void)n_in; (void)out_size; (void)ws_size;
  const int* value      = (const int*)d_in[0];
  const int* depth      = (const int*)d_in[1];
  const int* pos        = (const int*)d_in[2];
  const float* sos      = (const float*)d_in[3];
  const float* tok_emb  = (const float*)d_in[4];
  const float* depth_emb= (const float*)d_in[5];
  const float* pos_emb  = (const float*)d_in[6];
  const float* ln1_s    = (const float*)d_in[7];
  const float* ln1_b    = (const float*)d_in[8];
  const float* Wq       = (const float*)d_in[9];
  const float* Wk       = (const float*)d_in[10];
  const float* Wv       = (const float*)d_in[11];
  const float* Wo       = (const float*)d_in[12];
  const float* means    = (const float*)d_in[13];
  const float* ln2_s    = (const float*)d_in[14];
  const float* ln2_b    = (const float*)d_in[15];
  const float* W1       = (const float*)d_in[16];
  const float* b1       = (const float*)d_in[17];
  const float* W2       = (const float*)d_in[18];
  const float* b2       = (const float*)d_in[19];
  const float* head_w   = (const float*)d_in[20];

  const int M = MTOK;
  char* wsp = (char*)d_ws;
  auto alloc = [&](size_t bytes) {
    char* r = wsp; wsp += (bytes + 255) & ~(size_t)255; return r;
  };
  float*    x     = (float*)alloc((size_t)M * E_ * 4);
  float*    ab    = (float*)alloc((size_t)M * E_ * 4);
  float*    qn    = (float*)alloc((size_t)M * H_ * 4);
  float*    mn    = (float*)alloc((size_t)H_ * C_ * D_ * 4);
  float*    cnt   = (float*)alloc((size_t)B_ * H_ * S_ * 4);
  int*      idxb  = (int*)alloc((size_t)B_ * H_ * C_ * WIN_ * 4);
  _Float16* h16   = (_Float16*)alloc((size_t)M * E_ * 2);
  _Float16* q16   = (_Float16*)alloc((size_t)M * E_ * 2);
  _Float16* k16   = (_Float16*)alloc((size_t)M * E_ * 2);
  _Float16* v16   = (_Float16*)alloc((size_t)M * E_ * 2);
  _Float16* ab16  = (_Float16*)alloc((size_t)M * E_ * 2);
  _Float16* ff16  = (_Float16*)alloc((size_t)M * FF_ * 2);
  _Float16* x16   = (_Float16*)alloc((size_t)M * E_ * 2);
  _Float16* wq16  = (_Float16*)alloc((size_t)L_ * E_ * E_ * 2);
  _Float16* wk16  = (_Float16*)alloc((size_t)L_ * E_ * E_ * 2);
  _Float16* wv16  = (_Float16*)alloc((size_t)L_ * E_ * E_ * 2);
  _Float16* wo16  = (_Float16*)alloc((size_t)L_ * E_ * E_ * 2);
  _Float16* w116  = (_Float16*)alloc((size_t)L_ * E_ * FF_ * 2);
  _Float16* w216  = (_Float16*)alloc((size_t)L_ * FF_ * E_ * 2);
  _Float16* hd16  = (_Float16*)alloc((size_t)NVOC * E_ * 2);

  // one-time weight conversions (stateless: repeated every call)
  const int nWE = L_ * E_ * E_, nWF = L_ * E_ * FF_, nHD = NVOC * E_;
  f32_to_f16_kernel<<<(nWE + 255) / 256, 256, 0, stream>>>(Wq, wq16, nWE);
  f32_to_f16_kernel<<<(nWE + 255) / 256, 256, 0, stream>>>(Wk, wk16, nWE);
  f32_to_f16_kernel<<<(nWE + 255) / 256, 256, 0, stream>>>(Wv, wv16, nWE);
  f32_to_f16_kernel<<<(nWE + 255) / 256, 256, 0, stream>>>(Wo, wo16, nWE);
  f32_to_f16_kernel<<<(nWF + 255) / 256, 256, 0, stream>>>(W1, w116, nWF);
  f32_to_f16_kernel<<<(nWF + 255) / 256, 256, 0, stream>>>(W2, w216, nWF);
  f32_to_f16_kernel<<<(nHD + 255) / 256, 256, 0, stream>>>(head_w, hd16, nHD);

  embed_kernel<<<M, 256, 0, stream>>>(value, depth, pos, sos, tok_emb, depth_emb,
                                      pos_emb, x);

  dim3 gE(E_ / 64, M / 64);
  dim3 gF(FF_ / 64, M / 64);

  for (int l = 0; l < L_; ++l) {
    const _Float16* wq = wq16 + (size_t)l * E_ * E_;
    const _Float16* wk = wk16 + (size_t)l * E_ * E_;
    const _Float16* wv = wv16 + (size_t)l * E_ * E_;
    const _Float16* wo = wo16 + (size_t)l * E_ * E_;
    const _Float16* w1 = w116 + (size_t)l * E_ * FF_;
    const _Float16* w2 = w216 + (size_t)l * FF_ * E_;

    layernorm_kernel<<<M, 256, 0, stream>>>(x, ln1_s + (size_t)l * E_,
                                            ln1_b + (size_t)l * E_, h16);
    gemm_wmma_kernel<false, false, false, false, true><<<gE, 256, 0, stream>>>(
        h16, wq, nullptr, nullptr, q16, M, E_, E_);
    gemm_wmma_kernel<false, false, false, false, true><<<gE, 256, 0, stream>>>(
        h16, wk, nullptr, nullptr, k16, M, E_, E_);
    gemm_wmma_kernel<false, false, false, false, true><<<gE, 256, 0, stream>>>(
        h16, wv, nullptr, nullptr, v16, M, E_, E_);

    qnorm_kernel<<<M * H_ / 8, 256, 0, stream>>>(q16, qn);
    mnorm_kernel<<<H_ * C_, 64, 0, stream>>>(means + (size_t)l * H_ * C_ * D_, mn);
    route_topk_kernel<<<B_ * H_ * C_, 256, 0, stream>>>(q16, qn, mn, idxb);

    (void)hipMemsetAsync(ab, 0, (size_t)M * E_ * sizeof(float), stream);
    (void)hipMemsetAsync(cnt, 0, (size_t)B_ * H_ * S_ * sizeof(float), stream);
    bucket_attn_kernel<<<B_ * H_ * C_, 256, 0, stream>>>(q16, k16, v16, idxb,
                                                         value, ab, cnt);
    attn_norm_kernel<<<M, 256, 0, stream>>>(ab, cnt, ab16);

    // x = x + attn @ Wo
    gemm_wmma_kernel<false, false, true, false, false><<<gE, 256, 0, stream>>>(
        ab16, wo, nullptr, x, x, M, E_, E_);

    layernorm_kernel<<<M, 256, 0, stream>>>(x, ln2_s + (size_t)l * E_,
                                            ln2_b + (size_t)l * E_, h16);
    // ff = gelu(h @ W1 + b1)   (f16 out)
    gemm_wmma_kernel<false, true, false, true, true><<<gF, 256, 0, stream>>>(
        h16, w1, b1 + (size_t)l * FF_, nullptr, ff16, M, FF_, E_);
    // x = x + ff @ W2 + b2
    gemm_wmma_kernel<false, false, true, true, false><<<gE, 256, 0, stream>>>(
        ff16, w2, b2 + (size_t)l * E_, x, x, M, E_, FF_);
  }

  // logits = x @ head_w^T  (N = 17, guarded)
  const int nX = M * E_;
  f32_to_f16_kernel<<<(nX + 255) / 256, 256, 0, stream>>>(x, x16, nX);
  dim3 gH((NVOC + 63) / 64, M / 64);
  gemm_wmma_kernel<true, false, false, false, false><<<gH, 256, 0, stream>>>(
      x16, hd16, nullptr, nullptr, (float*)d_out, M, NVOC, E_);
}
